// Deform_Attn_80504866997033
// MI455X (gfx1250) — matrix operand order repaired and checked by reading, not compiled, not run on hardware
//
#include <hip/hip_runtime.h>
#include <hip/hip_bf16.h>

// ---------------------------------------------------------------------------
// Deformable attention decoder layer for MI455X (gfx1250, wave32, WMMA).
// All GEMMs run as bf16 WMMA (v_wmma_f32_16x16x32_bf16) with f32 accum.
// FFN is fused per 16-row tile with the 2048-wide hidden held in LDS.
// ---------------------------------------------------------------------------

#define D_MODEL 256
#define N_HEADS 8
#define N_POINTS 4
#define DIM_FF 2048
#define DH 32
#define B_ 2
#define LQ_ 16384
#define H_IMG 128
#define W_IMG 256
#define N_IN (H_IMG * W_IMG)
#define M_Q (B_ * LQ_)      // 32768 query rows
#define M_V (B_ * N_IN)     // 65536 value rows

typedef __attribute__((ext_vector_type(16))) __bf16 v16bf;
typedef __attribute__((ext_vector_type(8)))  float  v8f;

// ---- bf16 helpers ---------------------------------------------------------
__device__ __forceinline__ unsigned short f2bf(float x) {
    unsigned int u = __float_as_uint(x);
    u += 0x7fffu + ((u >> 16) & 1u);   // round-to-nearest-even
    return (unsigned short)(u >> 16);
}

// ---- WMMA fragment loaders (ISA 7.12.2 layouts, wave32) -------------------
// A (16x32 MxK, bf16): lane L -> m=L&15, hh=L>>4 ; VGPR v holds K pair
//   k = (v>=4?16:0) + hh*8 + (v&3)*2
__device__ __forceinline__ v16bf load_a_frag(const unsigned short* A, int lda) {
    const int lane = threadIdx.x & 31;
    const int m = lane & 15, hh = lane >> 4;
    const unsigned short* row = A + (size_t)m * lda;
    union { unsigned int u[8]; v16bf v; } f;
#pragma unroll
    for (int r = 0; r < 8; ++r) {
        const int k = ((r & 4) << 2) + hh * 8 + ((r & 3) << 1);
        f.u[r] = *(const unsigned int*)(row + k);
    }
    return f.v;
}

// B (32x16 KxN, bf16) with weight stored row-major W[n][k] (so B[k][n]=W[n][k]):
// lane L -> n=L&15, hh=L>>4 ; VGPR v holds K pair k = (v>=4?16:0) + hh*8 + (v&3)*2
__device__ __forceinline__ v16bf load_b_frag(const unsigned short* W, int ldw) {
    const int lane = threadIdx.x & 31;
    const int n = lane & 15, hh = lane >> 4;
    const unsigned short* row = W + (size_t)n * ldw + hh * 8;
    union { unsigned int u[8]; v16bf v; } f;
#pragma unroll
    for (int r = 0; r < 8; ++r) {
        f.u[r] = *(const unsigned int*)(row + ((r & 4) << 2) + ((r & 3) << 1));
    }
    return f.v;
}

__device__ __forceinline__ v8f wmma_bf16(v16bf a, v16bf b, v8f c) {
    return __builtin_amdgcn_wmma_f32_16x16x32_bf16(false, a, false, b,
                                                   (short)0, c, false, false);
}

// ---------------------------------------------------------------------------
// elementwise: out_bf16[i] = bf16(a[i] + (b ? b[i] : 0))
// ---------------------------------------------------------------------------
__global__ void cvt_bf16_kernel(const float* __restrict__ a,
                                const float* __restrict__ b,
                                unsigned short* __restrict__ o, int n) {
    int i = blockIdx.x * 256 + threadIdx.x;
    if (i < n) {
        float v = a[i] + (b ? b[i] : 0.0f);
        o[i] = f2bf(v);
    }
}

__global__ void concat_bias_kernel(const float* __restrict__ boff,
                                   const float* __restrict__ battn,
                                   float* __restrict__ bqk) {
    int i = threadIdx.x;
    if (i < 64) bqk[i] = boff[i];
    else if (i < 96) bqk[i] = battn[i - 64];
}

// ---------------------------------------------------------------------------
// Generic bf16 WMMA GEMM: C(MxN) = A(MxK) @ W(NxK)^T + bias
// block = 256 thr (8 waves); wave w -> rows [by*128 + w*16, +16),
// cols [bx*WN*16, +WN*16). Requires M%128==0, N%(WN*16)*gridx==N, K%32==0.
// ---------------------------------------------------------------------------
template <int WN>
__global__ void gemm_bf16_kernel(const unsigned short* __restrict__ A,
                                 const unsigned short* __restrict__ W,
                                 const float* __restrict__ bias,
                                 float* __restrict__ Cf,
                                 unsigned short* __restrict__ Cbf,
                                 int N, int K, int relu) {
    const int wave = threadIdx.x >> 5;
    const int lane = threadIdx.x & 31;
    const int row0 = blockIdx.y * 128 + wave * 16;
    const int col0 = blockIdx.x * (WN * 16);

    v8f acc[WN] = {};
    for (int k = 0; k < K; k += 32) {
        v16bf af = load_a_frag(A + (size_t)row0 * K + k, K);
#pragma unroll
        for (int t = 0; t < WN; ++t) {
            v16bf bf = load_b_frag(W + (size_t)(col0 + t * 16) * K + k, K);
            acc[t] = wmma_bf16(af, bf, acc[t]);
        }
    }

    const int nl = lane & 15, hh = lane >> 4;
#pragma unroll
    for (int t = 0; t < WN; ++t) {
        const int n = col0 + t * 16 + nl;
        const float bs = bias ? bias[n] : 0.0f;
#pragma unroll
        for (int r = 0; r < 8; ++r) {
            const int m = row0 + hh * 8 + r;
            float v = acc[t][r] + bs;
            if (relu) v = v > 0.0f ? v : 0.0f;
            const size_t idx = (size_t)m * N + n;
            if (Cf)  Cf[idx] = v;
            if (Cbf) Cbf[idx] = f2bf(v);
        }
    }
}

// ---------------------------------------------------------------------------
// Deformable sampling: thread = (b,q,h,d). Lanes span d -> coalesced gathers.
// V  : (B*N_IN, 256) f32 value-projected features (col = h*32 + d)
// QK : (M_Q, 96) f32 : cols 0..63 offsets (h,p,xy), 64..95 attn logits (h,p)
// out: bf16 (M_Q, 256) attention output, feeds out-proj WMMA GEMM
// ---------------------------------------------------------------------------
__global__ void sample_kernel(const float* __restrict__ V,
                              const float* __restrict__ QK,
                              const float* __restrict__ rp,
                              unsigned short* __restrict__ outbf) {
    const int tid = blockIdx.x * 256 + threadIdx.x;
    const int d  = tid & 31;
    const int h  = (tid >> 5) & 7;
    const int bq = tid >> 8;                    // 0 .. M_Q-1
    const int b  = bq >> 14;                    // LQ = 16384

    const float rx = rp[bq * 2 + 0];
    const float ry = rp[bq * 2 + 1];
    const float* row = QK + (size_t)bq * 96;

    // softmax over the 4 points of this head
    float lg[N_POINTS];
    float mx = -1e30f;
#pragma unroll
    for (int p = 0; p < N_POINTS; ++p) {
        lg[p] = row[64 + h * 4 + p];
        mx = fmaxf(mx, lg[p]);
    }
    float s = 0.0f;
#pragma unroll
    for (int p = 0; p < N_POINTS; ++p) { lg[p] = __expf(lg[p] - mx); s += lg[p]; }
    const float inv = 1.0f / s;

    const float* vb = V + (size_t)b * N_IN * 256 + h * 32 + d;
    float acc = 0.0f;
#pragma unroll
    for (int p = 0; p < N_POINTS; ++p) {
        const float ox = row[h * 8 + p * 2 + 0];
        const float oy = row[h * 8 + p * 2 + 1];
        // x = (rx + ox/W)*W - 0.5 ; y = (ry + oy/H)*H - 0.5
        const float x = rx * (float)W_IMG + ox - 0.5f;
        const float y = ry * (float)H_IMG + oy - 0.5f;
        const float x0 = floorf(x), y0 = floorf(y);
        const float wx1 = x - x0, wy1 = y - y0;
        const float wx0 = 1.0f - wx1, wy0 = 1.0f - wy1;
        const int x0i = (int)x0, y0i = (int)y0;

        float sv = 0.0f;
#pragma unroll
        for (int c = 0; c < 4; ++c) {
            const int xi = x0i + (c & 1);
            const int yi = y0i + (c >> 1);
            const float w = ((c & 1) ? wx1 : wx0) * ((c >> 1) ? wy1 : wy0);
            const bool valid = (xi >= 0) & (xi < W_IMG) & (yi >= 0) & (yi < H_IMG);
            const int xc = xi < 0 ? 0 : (xi > W_IMG - 1 ? W_IMG - 1 : xi);
            const int yc = yi < 0 ? 0 : (yi > H_IMG - 1 ? H_IMG - 1 : yi);
            const float v = vb[(size_t)(yc * W_IMG + xc) * 256];
            sv += valid ? w * v : 0.0f;
        }
        acc += lg[p] * inv * sv;
    }
    outbf[(size_t)bq * 256 + h * 32 + d] = f2bf(acc);
}

// ---------------------------------------------------------------------------
// LayerNorm over D=256: block = one row, 256 threads, LDS tree reduction.
// x = X[row] + (R ? R[row] : 0) ; out f32 and optional bf16.
// ---------------------------------------------------------------------------
__global__ void ln_kernel(const float* __restrict__ X,
                          const float* __restrict__ R,
                          const float* __restrict__ g,
                          const float* __restrict__ beta,
                          float* __restrict__ Of,
                          unsigned short* __restrict__ Obf) {
    const int row = blockIdx.x;
    const int i = threadIdx.x;
    const size_t idx = (size_t)row * 256 + i;
    float x = X[idx] + (R ? R[idx] : 0.0f);

    __shared__ float red[256];
    red[i] = x;
    __syncthreads();
#pragma unroll
    for (int s = 128; s > 0; s >>= 1) {
        if (i < s) red[i] += red[i + s];
        __syncthreads();
    }
    const float mu = red[0] * (1.0f / 256.0f);
    __syncthreads();
    const float dv = x - mu;
    red[i] = dv * dv;
    __syncthreads();
#pragma unroll
    for (int s = 128; s > 0; s >>= 1) {
        if (i < s) red[i] += red[i + s];
        __syncthreads();
    }
    const float var = red[0] * (1.0f / 256.0f);
    const float y = dv * rsqrtf(var + 1e-5f) * g[i] + beta[i];
    if (Of)  Of[idx] = y;
    if (Obf) Obf[idx] = f2bf(y);
}

// ---------------------------------------------------------------------------
// Fused FFN: y = relu(t @ W1^T + b1) @ W2^T + b2 + t   (per 16-row tile)
// block = 256 thr (8 waves), grid = M/16. Hidden kept in LDS (2 chunks of
// 1024 cols, 32KB bf16). Each wave: FFN1 -> 8 n-tiles, FFN2 -> 2 n-tiles.
// ---------------------------------------------------------------------------
__global__ void ffn_fused_kernel(const unsigned short* __restrict__ Tbf,
                                 const float* __restrict__ Tf32,
                                 const unsigned short* __restrict__ W1,
                                 const float* __restrict__ b1,
                                 const unsigned short* __restrict__ W2,
                                 const float* __restrict__ b2,
                                 float* __restrict__ Y) {
    __shared__ unsigned short hid[16 * 1024];   // 32 KB bf16 hidden chunk
    const int wave = threadIdx.x >> 5;
    const int lane = threadIdx.x & 31;
    const int row0 = blockIdx.x * 16;
    const int nl = lane & 15, hh = lane >> 4;

    v8f acc2[2] = {};
#pragma unroll
    for (int chunk = 0; chunk < 2; ++chunk) {
        // ---- FFN1: hidden cols [chunk*1024 + wave*128, +128) ----
        const int hc0 = chunk * 1024 + wave * 128;
        v8f acc1[8] = {};
        for (int k = 0; k < 256; k += 32) {
            v16bf af = load_a_frag(Tbf + (size_t)row0 * 256 + k, 256);
#pragma unroll
            for (int t = 0; t < 8; ++t) {
                v16bf bf = load_b_frag(W1 + (size_t)(hc0 + t * 16) * 256 + k, 256);
                acc1[t] = wmma_bf16(af, bf, acc1[t]);
            }
        }
        __syncthreads();   // chunk1: previous FFN2 LDS reads complete (WAR)
        // bias + relu + bf16 -> LDS
#pragma unroll
        for (int t = 0; t < 8; ++t) {
            const int ncol = hc0 + t * 16 + nl;
            const int lcol = wave * 128 + t * 16 + nl;
            const float bs = b1[ncol];
#pragma unroll
            for (int r = 0; r < 8; ++r) {
                const int m = hh * 8 + r;
                float v = acc1[t][r] + bs;
                v = v > 0.0f ? v : 0.0f;
                hid[m * 1024 + lcol] = f2bf(v);
            }
        }
        __syncthreads();   // hidden visible to all waves
        // ---- FFN2: accumulate over this 1024-wide K chunk ----
        for (int k = 0; k < 1024; k += 32) {
            v16bf af = load_a_frag(&hid[0] + k, 1024);   // LDS A fragment
#pragma unroll
            for (int t = 0; t < 2; ++t) {
                v16bf bf = load_b_frag(
                    W2 + (size_t)(wave * 32 + t * 16) * 2048 + chunk * 1024 + k,
                    2048);
                acc2[t] = wmma_bf16(af, bf, acc2[t]);
            }
        }
    }

    // epilogue: + b2 + residual t
#pragma unroll
    for (int t = 0; t < 2; ++t) {
        const int n = wave * 32 + t * 16 + nl;
        const float bs = b2[n];
#pragma unroll
        for (int r = 0; r < 8; ++r) {
            const int m = row0 + hh * 8 + r;
            const size_t idx = (size_t)m * 256 + n;
            Y[idx] = acc2[t][r] + bs + Tf32[idx];
        }
    }
}

// ---------------------------------------------------------------------------
extern "C" void kernel_launch(void* const* d_in, const int* in_sizes, int n_in,
                              void* d_out, int out_size, void* d_ws, size_t ws_size,
                              hipStream_t stream) {
    (void)in_sizes; (void)n_in; (void)out_size; (void)ws_size;
    const float* tgt    = (const float*)d_in[0];
    const float* uv     = (const float*)d_in[1];
    const float* rp     = (const float*)d_in[2];
    const float* qpos   = (const float*)d_in[3];
    const float* W_off  = (const float*)d_in[4];
    const float* b_off  = (const float*)d_in[5];
    const float* W_attn = (const float*)d_in[6];
    const float* b_attn = (const float*)d_in[7];
    const float* W_val  = (const float*)d_in[8];
    const float* b_val  = (const float*)d_in[9];
    const float* W_out  = (const float*)d_in[10];
    const float* b_out  = (const float*)d_in[11];
    const float* W1     = (const float*)d_in[12];
    const float* b1     = (const float*)d_in[13];
    const float* W2     = (const float*)d_in[14];
    const float* b2     = (const float*)d_in[15];
    const float* g1     = (const float*)d_in[16];
    const float* beta1  = (const float*)d_in[17];
    const float* g2     = (const float*)d_in[18];
    const float* beta2  = (const float*)d_in[19];
    float* out = (float*)d_out;

    // ---- workspace carve-up (256B aligned) ----
    char* ws = (char*)d_ws;
    size_t off = 0;
    auto alloc = [&](size_t bytes) -> char* {
        char* p = ws + off;
        off = (off + bytes + 255) & ~(size_t)255;
        return p;
    };
    unsigned short* uv_bf   = (unsigned short*)alloc((size_t)M_V * 256 * 2);
    unsigned short* q_bf    = (unsigned short*)alloc((size_t)M_Q * 256 * 2);
    unsigned short* Wval_bf = (unsigned short*)alloc(256 * 256 * 2);
    unsigned short* Wqk_bf  = (unsigned short*)alloc(96 * 256 * 2);
    unsigned short* Wout_bf = (unsigned short*)alloc(256 * 256 * 2);
    unsigned short* W1_bf   = (unsigned short*)alloc(2048 * 256 * 2);
    unsigned short* W2_bf   = (unsigned short*)alloc(256 * 2048 * 2);
    float*          bqk     = (float*)alloc(96 * 4);
    float*          V       = (float*)alloc((size_t)M_V * 256 * 4);
    float*          QK      = (float*)alloc((size_t)M_Q * 96 * 4);
    unsigned short* att_bf  = (unsigned short*)alloc((size_t)M_Q * 256 * 2);
    float*          tgt2    = (float*)alloc((size_t)M_Q * 256 * 4);
    float*          Tf32    = (float*)alloc((size_t)M_Q * 256 * 4);
    unsigned short* Tbf     = (unsigned short*)alloc((size_t)M_Q * 256 * 2);
    float*          Y       = (float*)alloc((size_t)M_Q * 256 * 4);

    auto blocks = [](size_t n) { return (unsigned)((n + 255) / 256); };

    // ---- 1. bf16 conversions ----
    cvt_bf16_kernel<<<blocks((size_t)M_V * 256), 256, 0, stream>>>(uv, nullptr, uv_bf, M_V * 256);
    cvt_bf16_kernel<<<blocks((size_t)M_Q * 256), 256, 0, stream>>>(tgt, qpos, q_bf, M_Q * 256);
    cvt_bf16_kernel<<<blocks(256 * 256), 256, 0, stream>>>(W_val, nullptr, Wval_bf, 256 * 256);
    cvt_bf16_kernel<<<blocks(64 * 256), 256, 0, stream>>>(W_off, nullptr, Wqk_bf, 64 * 256);
    cvt_bf16_kernel<<<blocks(32 * 256), 256, 0, stream>>>(W_attn, nullptr, Wqk_bf + 64 * 256, 32 * 256);
    cvt_bf16_kernel<<<blocks(256 * 256), 256, 0, stream>>>(W_out, nullptr, Wout_bf, 256 * 256);
    cvt_bf16_kernel<<<blocks(2048 * 256), 256, 0, stream>>>(W1, nullptr, W1_bf, 2048 * 256);
    cvt_bf16_kernel<<<blocks(256 * 2048), 256, 0, stream>>>(W2, nullptr, W2_bf, 256 * 2048);
    concat_bias_kernel<<<1, 96, 0, stream>>>(b_off, b_attn, bqk);

    // ---- 2. value projection: V = uv @ W_val^T + b_val  (65536x256) ----
    gemm_bf16_kernel<4><<<dim3(256 / 64, M_V / 128), 256, 0, stream>>>(
        uv_bf, Wval_bf, b_val, V, nullptr, 256, 256, 0);

    // ---- 3. offsets + attn logits: QK = q @ [W_off;W_attn]^T + bqk ----
    gemm_bf16_kernel<2><<<dim3(96 / 32, M_Q / 128), 256, 0, stream>>>(
        q_bf, Wqk_bf, bqk, QK, nullptr, 96, 256, 0);

    // ---- 4. deformable sampling (softmax fused) -> att_bf ----
    sample_kernel<<<(M_Q * N_HEADS * DH) / 256, 256, 0, stream>>>(V, QK, rp, att_bf);

    // ---- 5. out projection: tgt2 = att @ W_out^T + b_out ----
    gemm_bf16_kernel<4><<<dim3(256 / 64, M_Q / 128), 256, 0, stream>>>(
        att_bf, Wout_bf, b_out, tgt2, nullptr, 256, 256, 0);

    // ---- 6. LN1: t = LN(tgt + tgt2) -> f32 + bf16 ----
    ln_kernel<<<M_Q, 256, 0, stream>>>(tgt2, tgt, g1, beta1, Tf32, Tbf);

    // ---- 7. fused FFN: Y = relu(t@W1^T+b1)@W2^T + b2 + t ----
    ffn_fused_kernel<<<M_Q / 16, 256, 0, stream>>>(Tbf, Tf32, W1_bf, b1, W2_bf, b2, Y);

    // ---- 8. LN2 -> d_out ----
    ln_kernel<<<M_Q, 256, 0, stream>>>(Y, nullptr, g2, beta2, out, nullptr);
}
